// FeedForward_6863357739124
// MI455X (gfx1250) — compile-verified, matrix operand order checked by reading, not run on hardware
//
#include <hip/hip_runtime.h>

typedef __bf16 bf16_t;
typedef __attribute__((ext_vector_type(16))) __bf16 v16bf;
typedef __attribute__((ext_vector_type(8)))  float  v8f;
typedef __attribute__((ext_vector_type(4)))  unsigned int u32x4;
typedef __attribute__((ext_vector_type(4)))  int i32x4;
typedef __attribute__((ext_vector_type(8)))  int i32x8;

#if defined(__has_builtin)
#if __has_builtin(__builtin_amdgcn_tensor_load_to_lds)
#define HAVE_TDM 1
#endif
#endif
#ifndef HAVE_TDM
#define HAVE_TDM 0
#endif

#if HAVE_TDM
#define NSTAGE 2
#else
#define NSTAGE 1
#endif

// Tile geometry: BM=BN=128 rows, BK=32 K-elements (one WMMA K-step per stage).
// LDS row = 64B data + 16B pad (TDM pad_interval=16DW code 3, pad_amount=4DW code 3)
// -> stride 80B keeps ds_load_b128 16B-aligned and spreads 16 fragment rows over
//    distinct bank quads (20-bank stride covers all 64 banks in 16 rows).
#define BK 32
#define TILE_ROWS 128
#define ROW_BYTES 80
#define TILE_BYTES (TILE_ROWS * ROW_BYTES)   // 10240

// ---------------- TDM descriptor helpers (ISA 08_async_tensor §8) ----------------

__device__ __forceinline__ i32x8 tdm_group1(unsigned dim0, unsigned dim1) {
  // [17:16]=data_size(2B)=1, [20]=pad_enable, [24:22]=pad_interval code 3 (16 DW),
  // [31:25]=pad_amount code 3 (4 DW); tensor_dim0 @48, tensor_dim1 @80,
  // tile_dim0 @112 = BK, tile_dim1 @128 = TILE_ROWS, tensor_dim0_stride @160 = dim0.
  i32x8 g;
  g[0] = (int)((1u << 16) | (1u << 20) | (3u << 22) | (3u << 25));
  g[1] = (int)((dim0 & 0xFFFFu) << 16);
  g[2] = (int)(((dim0 >> 16) & 0xFFFFu) | ((dim1 & 0xFFFFu) << 16));
  g[3] = (int)(((dim1 >> 16) & 0xFFFFu) | ((unsigned)BK << 16));
  g[4] = (int)(TILE_ROWS & 0xFFFFu);
  g[5] = (int)dim0;
  g[6] = 0;
  g[7] = 0;
  return g;
}

__device__ __forceinline__ void tdm_load_tile(const void* gaddr, unsigned ldsOff, i32x8 g1) {
  unsigned long long ga = (unsigned long long)gaddr;
  u32x4 g0;
  g0[0] = 1u;                                    // count=1, user descriptor
  g0[1] = ldsOff;                                // lds byte address
  g0[2] = (unsigned)ga;                          // global_addr[31:0]
  g0[3] = (unsigned)((ga >> 32) & 0x01FFFFFFu) | 0x80000000u;  // ga[56:32], type=2
  i32x4 z4 = {0, 0, 0, 0};
#if HAVE_TDM
#if __clang_major__ >= 23
  i32x8 z8 = {0, 0, 0, 0, 0, 0, 0, 0};
  __builtin_amdgcn_tensor_load_to_lds(g0, g1, z4, z4, z8, 0);
#else
  __builtin_amdgcn_tensor_load_to_lds(g0, g1, z4, z4, 0);
#endif
#else
  (void)g0; (void)g1; (void)z4;
#endif
}

// Fallback cooperative fill (no padding semantics difference: we write same layout)
__device__ __forceinline__ void fill_tile_sync(char* dst, const bf16_t* g, int ldK) {
  for (int c = threadIdx.x; c < TILE_ROWS * 4; c += 256) {
    int row = c >> 2;
    int col = (c & 3) * 16;
    uint4 v = *reinterpret_cast<const uint4*>((const char*)g + (size_t)row * ldK * 2 + col);
    *reinterpret_cast<uint4*>(dst + (size_t)row * ROW_BYTES + col) = v;
  }
}

// ---------------- fragment load from LDS ----------------
// 16-bit A/B fragment: lanes 0-15 row r, K[0..7]+[16..23]; lanes 16-31 row r, K[8..15]+[24..31]

__device__ __forceinline__ v16bf ld_frag(const char* p) {
  union { v16bf v; uint4 q[2]; } f;
  f.q[0] = *reinterpret_cast<const uint4*>(p);
  f.q[1] = *reinterpret_cast<const uint4*>(p + 32);
  return f.v;
}

// ---------------- IF spiking rate (T=10, threshold 1.0, hard reset) ----------------

__device__ __forceinline__ float if_rate(float c) {
  float v = 0.0f, s = 0.0f;
#pragma unroll
  for (int t = 0; t < 10; ++t) {
    float h  = v + c;
    float sp = (h >= 1.0f) ? 1.0f : 0.0f;
    v = h * (1.0f - sp);
    s += sp;
  }
  return s * 0.1f;
}

// ---------------- fused layer-1: c1=Xp*W1^T, c3p=Xp*W3^T, c3n=Xn*W3^T,
//                  p = rate(c1) * (rate(c3p) - rate(c3n))  [bf16 out] ----------------

__global__ __launch_bounds__(256) void ff_layer1_fused(
    const bf16_t* __restrict__ xpos, const bf16_t* __restrict__ xneg,
    const bf16_t* __restrict__ w3, const bf16_t* __restrict__ w1,
    bf16_t* __restrict__ pout, int M, int Nout, int K) {
  __shared__ __align__(128) char smem[NSTAGE * 4 * TILE_BYTES];

  const int lane  = threadIdx.x & 31;
  const int wid   = threadIdx.x >> 5;
  const int wm    = wid >> 2;          // 0..1  (64-row slab)
  const int wn    = wid & 3;           // 0..3  (32-col slab)
  const int m0    = blockIdx.y * 128;
  const int n0    = blockIdx.x * 128;
  const int r     = lane & 15;
  const int koffB = (lane < 16) ? 0 : 16;   // bytes into the 64B row chunk
  const unsigned ldsBase = (unsigned)(unsigned long long)(void*)smem;

  const v8f zero = {0.f, 0.f, 0.f, 0.f, 0.f, 0.f, 0.f, 0.f};
  v8f acc1[4][2], acc3p[4][2], acc3n[4][2];
#pragma unroll
  for (int i = 0; i < 4; ++i)
#pragma unroll
    for (int j = 0; j < 2; ++j) { acc1[i][j] = zero; acc3p[i][j] = zero; acc3n[i][j] = zero; }

  const int nStages = K / BK;

  auto compute = [&](int buf) {
    const char* base = smem + (size_t)buf * (4 * TILE_BYTES);
    v16bf axp[4], axn[4], wf3[2], wf1[2];
#pragma unroll
    for (int i = 0; i < 4; ++i) {
      const char* pa = base + (size_t)(wm * 64 + 16 * i + r) * ROW_BYTES + koffB;
      axp[i] = ld_frag(pa);
      axn[i] = ld_frag(pa + TILE_BYTES);
    }
#pragma unroll
    for (int j = 0; j < 2; ++j) {
      const char* pw = base + 2 * TILE_BYTES + (size_t)(wn * 32 + 16 * j + r) * ROW_BYTES + koffB;
      wf3[j] = ld_frag(pw);
      wf1[j] = ld_frag(pw + TILE_BYTES);
    }
#pragma unroll
    for (int i = 0; i < 4; ++i)
#pragma unroll
      for (int j = 0; j < 2; ++j) {
        acc1[i][j]  = __builtin_amdgcn_wmma_f32_16x16x32_bf16(false, axp[i], false, wf1[j], (short)0, acc1[i][j],  false, false);
        acc3p[i][j] = __builtin_amdgcn_wmma_f32_16x16x32_bf16(false, axp[i], false, wf3[j], (short)0, acc3p[i][j], false, false);
        acc3n[i][j] = __builtin_amdgcn_wmma_f32_16x16x32_bf16(false, axn[i], false, wf3[j], (short)0, acc3n[i][j], false, false);
      }
  };

#if HAVE_TDM
  const i32x8 gA = tdm_group1((unsigned)K, (unsigned)M);
  const i32x8 gW = tdm_group1((unsigned)K, (unsigned)Nout);
  auto issue = [&](int s) {
    unsigned buf = (unsigned)(s & 1) * (4 * TILE_BYTES);
    size_t kb = (size_t)s * BK * 2;   // byte offset along K
    tdm_load_tile((const char*)xpos + (size_t)m0 * K * 2 + kb, ldsBase + buf + 0 * TILE_BYTES, gA);
    tdm_load_tile((const char*)xneg + (size_t)m0 * K * 2 + kb, ldsBase + buf + 1 * TILE_BYTES, gA);
    tdm_load_tile((const char*)w3   + (size_t)n0 * K * 2 + kb, ldsBase + buf + 2 * TILE_BYTES, gW);
    tdm_load_tile((const char*)w1   + (size_t)n0 * K * 2 + kb, ldsBase + buf + 3 * TILE_BYTES, gW);
  };
  if (wid == 0) issue(0);
  for (int s = 0; s < nStages; ++s) {
    if (wid == 0) {
      if (s + 1 < nStages) { issue(s + 1); __builtin_amdgcn_s_wait_tensorcnt(4); }
      else                 { __builtin_amdgcn_s_wait_tensorcnt(0); }
    }
    __syncthreads();          // stage s visible to all waves
    compute(s & 1);
    __syncthreads();          // all reads done before buffer reuse
  }
#else
  for (int s = 0; s < nStages; ++s) {
    size_t kel = (size_t)s * BK;
    fill_tile_sync(smem + 0 * TILE_BYTES, xpos + (size_t)m0 * K + kel, K);
    fill_tile_sync(smem + 1 * TILE_BYTES, xneg + (size_t)m0 * K + kel, K);
    fill_tile_sync(smem + 2 * TILE_BYTES, w3   + (size_t)n0 * K + kel, K);
    fill_tile_sync(smem + 3 * TILE_BYTES, w1   + (size_t)n0 * K + kel, K);
    __syncthreads();
    compute(0);
    __syncthreads();
  }
#endif

  // Epilogue: spike rates on registers, write bf16 p.
  const int rowAdd = (lane < 16) ? 0 : 8;
#pragma unroll
  for (int i = 0; i < 4; ++i) {
#pragma unroll
    for (int j = 0; j < 2; ++j) {
      int gr = m0 + wm * 64 + 16 * i + rowAdd;
      int gc = n0 + wn * 32 + 16 * j + (lane & 15);
#pragma unroll
      for (int v = 0; v < 8; ++v) {
        float r1 = if_rate(acc1[i][j][v]);
        float r3 = if_rate(acc3p[i][j][v]) - if_rate(acc3n[i][j][v]);
        pout[(size_t)(gr + v) * Nout + gc] = (bf16_t)(r1 * r3);
      }
    }
  }
}

// ---------------- plain NT GEMM (layer 2): C[M,N] = A[M,K] * W[N,K]^T ----------------

__global__ __launch_bounds__(256) void gemm_nt_tdm(
    const bf16_t* __restrict__ A, const bf16_t* __restrict__ W,
    float* __restrict__ C, int M, int Nout, int K) {
  __shared__ __align__(128) char smem[NSTAGE * 2 * TILE_BYTES];

  const int lane  = threadIdx.x & 31;
  const int wid   = threadIdx.x >> 5;
  const int wm    = wid >> 2;
  const int wn    = wid & 3;
  const int m0    = blockIdx.y * 128;
  const int n0    = blockIdx.x * 128;
  const int r     = lane & 15;
  const int koffB = (lane < 16) ? 0 : 16;
  const unsigned ldsBase = (unsigned)(unsigned long long)(void*)smem;

  const v8f zero = {0.f, 0.f, 0.f, 0.f, 0.f, 0.f, 0.f, 0.f};
  v8f acc[4][2];
#pragma unroll
  for (int i = 0; i < 4; ++i)
#pragma unroll
    for (int j = 0; j < 2; ++j) acc[i][j] = zero;

  const int nStages = K / BK;

  auto compute = [&](int buf) {
    const char* base = smem + (size_t)buf * (2 * TILE_BYTES);
    v16bf af[4], wf[2];
#pragma unroll
    for (int i = 0; i < 4; ++i)
      af[i] = ld_frag(base + (size_t)(wm * 64 + 16 * i + r) * ROW_BYTES + koffB);
#pragma unroll
    for (int j = 0; j < 2; ++j)
      wf[j] = ld_frag(base + TILE_BYTES + (size_t)(wn * 32 + 16 * j + r) * ROW_BYTES + koffB);
#pragma unroll
    for (int i = 0; i < 4; ++i)
#pragma unroll
      for (int j = 0; j < 2; ++j)
        acc[i][j] = __builtin_amdgcn_wmma_f32_16x16x32_bf16(false, af[i], false, wf[j], (short)0, acc[i][j], false, false);
  };

#if HAVE_TDM
  const i32x8 gA = tdm_group1((unsigned)K, (unsigned)M);
  const i32x8 gW = tdm_group1((unsigned)K, (unsigned)Nout);
  auto issue = [&](int s) {
    unsigned buf = (unsigned)(s & 1) * (2 * TILE_BYTES);
    size_t kb = (size_t)s * BK * 2;
    tdm_load_tile((const char*)A + (size_t)m0 * K * 2 + kb, ldsBase + buf + 0 * TILE_BYTES, gA);
    tdm_load_tile((const char*)W + (size_t)n0 * K * 2 + kb, ldsBase + buf + 1 * TILE_BYTES, gW);
  };
  if (wid == 0) issue(0);
  for (int s = 0; s < nStages; ++s) {
    if (wid == 0) {
      if (s + 1 < nStages) { issue(s + 1); __builtin_amdgcn_s_wait_tensorcnt(2); }
      else                 { __builtin_amdgcn_s_wait_tensorcnt(0); }
    }
    __syncthreads();
    compute(s & 1);
    __syncthreads();
  }
#else
  for (int s = 0; s < nStages; ++s) {
    size_t kel = (size_t)s * BK;
    fill_tile_sync(smem + 0 * TILE_BYTES, A + (size_t)m0 * K + kel, K);
    fill_tile_sync(smem + 1 * TILE_BYTES, W + (size_t)n0 * K + kel, K);
    __syncthreads();
    compute(0);
    __syncthreads();
  }
#endif

  const int rowAdd = (lane < 16) ? 0 : 8;
#pragma unroll
  for (int i = 0; i < 4; ++i) {
#pragma unroll
    for (int j = 0; j < 2; ++j) {
      float* cp = C + (size_t)(m0 + wm * 64 + 16 * i + rowAdd) * Nout + (n0 + wn * 32 + 16 * j + (lane & 15));
#pragma unroll
      for (int v = 0; v < 8; ++v) cp[(size_t)v * Nout] = acc[i][j][v];
    }
  }
}

// ---------------- elementwise prep (vectorized x4) ----------------

union BF4 { bf16_t h[4]; uint2 u; };

__global__ __launch_bounds__(256) void relu_split_bf16_v4(const float4* __restrict__ x,
                                                          uint2* __restrict__ xpos,
                                                          uint2* __restrict__ xneg, int n4) {
  int i = blockIdx.x * 256 + threadIdx.x;
  if (i < n4) {
    float4 v = x[i];
    BF4 p, q;
    p.h[0] = (bf16_t)fmaxf(v.x, 0.f);  q.h[0] = (bf16_t)fmaxf(-v.x, 0.f);
    p.h[1] = (bf16_t)fmaxf(v.y, 0.f);  q.h[1] = (bf16_t)fmaxf(-v.y, 0.f);
    p.h[2] = (bf16_t)fmaxf(v.z, 0.f);  q.h[2] = (bf16_t)fmaxf(-v.z, 0.f);
    p.h[3] = (bf16_t)fmaxf(v.w, 0.f);  q.h[3] = (bf16_t)fmaxf(-v.w, 0.f);
    xpos[i] = p.u;
    xneg[i] = q.u;
  }
}

__global__ __launch_bounds__(256) void cvt_bf16_v4(const float4* __restrict__ w,
                                                   uint2* __restrict__ wb, int n4) {
  int i = blockIdx.x * 256 + threadIdx.x;
  if (i < n4) {
    float4 v = w[i];
    BF4 o;
    o.h[0] = (bf16_t)v.x; o.h[1] = (bf16_t)v.y; o.h[2] = (bf16_t)v.z; o.h[3] = (bf16_t)v.w;
    wb[i] = o.u;
  }
}

// ---------------- launcher ----------------

extern "C" void kernel_launch(void* const* d_in, const int* in_sizes, int n_in,
                              void* d_out, int out_size, void* d_ws, size_t ws_size,
                              hipStream_t stream) {
  const float* x  = (const float*)d_in[0];  // [N, D]
  const float* w1 = (const float*)d_in[1];  // [H, D]
  const float* w2 = (const float*)d_in[2];  // [D, H]
  const float* w3 = (const float*)d_in[3];  // [H, D]
  float* out = (float*)d_out;               // [N, D]

  const int N = 1024, D = 2048, H = 7168;

  char* ws = (char*)d_ws;
  size_t off = 0;
  bf16_t* xpos = (bf16_t*)(ws + off); off += (size_t)N * D * 2;
  bf16_t* xneg = (bf16_t*)(ws + off); off += (size_t)N * D * 2;
  bf16_t* w1b  = (bf16_t*)(ws + off); off += (size_t)H * D * 2;
  bf16_t* w3b  = (bf16_t*)(ws + off); off += (size_t)H * D * 2;
  bf16_t* w2b  = (bf16_t*)(ws + off); off += (size_t)D * H * 2;
  bf16_t* pb   = (bf16_t*)(ws + off); off += (size_t)N * H * 2;

  relu_split_bf16_v4<<<(N * D / 4 + 255) / 256, 256, 0, stream>>>(
      (const float4*)x, (uint2*)xpos, (uint2*)xneg, N * D / 4);
  cvt_bf16_v4<<<(H * D / 4 + 255) / 256, 256, 0, stream>>>((const float4*)w1, (uint2*)w1b, H * D / 4);
  cvt_bf16_v4<<<(H * D / 4 + 255) / 256, 256, 0, stream>>>((const float4*)w3, (uint2*)w3b, H * D / 4);
  cvt_bf16_v4<<<(D * H / 4 + 255) / 256, 256, 0, stream>>>((const float4*)w2, (uint2*)w2b, D * H / 4);

  // fused layer-1 (3 GEMMs + IF spiking) -> p [N, H] bf16
  dim3 g1(H / 128, N / 128);
  ff_layer1_fused<<<g1, 256, 0, stream>>>(xpos, xneg, w3b, w1b, pb, N, H, D);

  // layer-2: out = p @ w2^T  [N, D] f32
  dim3 g2(D / 128, N / 128);
  gemm_nt_tdm<<<g2, 256, 0, stream>>>(pb, w2b, out, N, D, H);
}
